// MultiheadAttention_53558242181498
// MI455X (gfx1250) — matrix operand order repaired and checked by reading, not compiled, not run on hardware
//
#include <hip/hip_runtime.h>
#include <hip/hip_bf16.h>

typedef __bf16 bf16x16 __attribute__((ext_vector_type(16)));
typedef float f32x8 __attribute__((ext_vector_type(8)));
typedef unsigned int u32x4 __attribute__((ext_vector_type(4)));

union FragAB {
  bf16x16 v;
  u32x4 q[2];
};

__device__ __forceinline__ unsigned short f2bf(float f) {
  unsigned int u = __builtin_bit_cast(unsigned int, f);
  u += 0x7FFFu + ((u >> 16) & 1u);  // round-to-nearest-even
  return (unsigned short)(u >> 16);
}

#define EMBED 768
#define SEQ 2048
#define NB 2
#define NH 12
#define HD 64
#define MTOT (NB * SEQ) /* 4096 */
#define KU (EMBED / 2)  /* 384 uints per row */

// ---------------- x (f32) -> bf16, 2-wide packed ----------------
__global__ void cvt_x_kernel(const float2* __restrict__ x,
                             unsigned int* __restrict__ y, int n2) {
  int i = blockIdx.x * blockDim.x + threadIdx.x;
  if (i < n2) {
    float2 v = x[i];
    y[i] = (unsigned int)f2bf(v.x) | ((unsigned int)f2bf(v.y) << 16);
  }
}

// ---------------- W (f32, [k][n]) -> Wt (bf16, [n][k]) tiled transpose ----------------
__global__ __launch_bounds__(256) void cvt_wt_kernel(
    const float* __restrict__ W, unsigned short* __restrict__ Wt) {
  __shared__ unsigned short t[32][33];
  const int bn = blockIdx.x * 32;  // n base
  const int bk = blockIdx.y * 32;  // k base
  const int tx = threadIdx.x & 31, ty = threadIdx.x >> 5;  // 32 x 8
#pragma unroll
  for (int r = 0; r < 4; ++r)
    t[ty + 8 * r][tx] = f2bf(W[(bk + ty + 8 * r) * EMBED + bn + tx]);
  __syncthreads();
#pragma unroll
  for (int r = 0; r < 4; ++r)
    Wt[(size_t)(bn + ty + 8 * r) * EMBED + bk + tx] = t[tx][ty + 8 * r];
}

// ---------------- GEMM: C[4096,768] = Abf16[4096,768] * Bt_bf16[768,768]^T ----------------
// Bt is [n][k].  MODE 0: bf16 head-layout output; MODE 1: f32 row-major output.
template <int MODE>
__global__ __launch_bounds__(256) void gemm_kernel(
    const unsigned short* __restrict__ A, const unsigned short* __restrict__ Bt,
    unsigned short* __restrict__ obf, float* __restrict__ of32) {
  __shared__ __align__(16) unsigned short Alds[2][128 * 40];  // [m][kk], pitch 40
  __shared__ __align__(16) unsigned short Blds[2][128 * 40];  // [n][kk], pitch 40
  const int tid = threadIdx.x;
  const int lane = tid & 31;
  const int w = tid >> 5;
  const int wm = w >> 2, wn = w & 3;
  const int m0 = blockIdx.y * 128, n0 = blockIdx.x * 128;
  const int l15 = lane & 15;
  const int hi = (lane >> 4) & 1;

  const unsigned int* Ag = (const unsigned int*)A;
  const unsigned int* Bg = (const unsigned int*)Bt;

  auto stage = [&](int buf, int k0) {
    unsigned int* Au = (unsigned int*)Alds[buf];
    unsigned int* Bu = (unsigned int*)Blds[buf];
#pragma unroll
    for (int e = 0; e < 8; ++e) {
      int idx = e * 256 + tid;
      int r = idx >> 4, ui = idx & 15;
      Au[r * 20 + ui] = Ag[(size_t)(m0 + r) * KU + (k0 >> 1) + ui];
      Bu[r * 20 + ui] = Bg[(size_t)(n0 + r) * KU + (k0 >> 1) + ui];
    }
  };

  f32x8 acc[4][2] = {};
  stage(0, 0);
  int buf = 0;
  for (int k0 = 0; k0 < EMBED; k0 += 32) {
    __syncthreads();
    if (k0 + 32 < EMBED) {
      stage(buf ^ 1, k0 + 32);
      if (k0 + 64 < EMBED) {  // prefetch tile after next (global_prefetch_b8)
        int r = tid >> 1;
        __builtin_prefetch(&Ag[(size_t)(m0 + r) * KU + ((k0 + 64) >> 1)], 0, 0);
        __builtin_prefetch(&Bg[(size_t)(n0 + r) * KU + ((k0 + 64) >> 1)], 0, 0);
      }
    }
    const unsigned short* Ab = Alds[buf];
    const unsigned short* Bb = Blds[buf];

    FragAB af[4];
#pragma unroll
    for (int mi = 0; mi < 4; ++mi) {
      int off = (wm * 64 + mi * 16 + l15) * 40 + hi * 8;
      af[mi].q[0] = *(const u32x4*)(const void*)&Ab[off];
      af[mi].q[1] = *(const u32x4*)(const void*)&Ab[off + 16];
    }
#pragma unroll
    for (int ni = 0; ni < 2; ++ni) {
      int off = (wn * 32 + ni * 16 + l15) * 40 + hi * 8;
      FragAB bfr;
      bfr.q[0] = *(const u32x4*)(const void*)&Bb[off];
      bfr.q[1] = *(const u32x4*)(const void*)&Bb[off + 16];
#pragma unroll
      for (int mi = 0; mi < 4; ++mi)
        acc[mi][ni] = __builtin_amdgcn_wmma_f32_16x16x32_bf16(
            false, af[mi].v, false, bfr.v, (short)0, acc[mi][ni], false, false);
    }
    buf ^= 1;
  }

  const int rowoff = hi * 8;
#pragma unroll
  for (int mi = 0; mi < 4; ++mi)
#pragma unroll
    for (int ni = 0; ni < 2; ++ni)
#pragma unroll
      for (int vi = 0; vi < 8; ++vi) {
        int m = m0 + wm * 64 + mi * 16 + rowoff + vi;
        int n = n0 + wn * 32 + ni * 16 + l15;
        float v = acc[mi][ni][vi];
        if (MODE == 0) {
          int b = m >> 11, s = m & 2047, h = n >> 6, hd = n & 63;
          obf[(((size_t)(b * NH + h) << 11) + s) * HD + hd] = f2bf(v);
        } else {
          of32[(size_t)m * EMBED + n] = v;
        }
      }
}

// ---------------- Flash attention: block = one (b,h) x 128 query rows, 8 waves ----------------
__global__ __launch_bounds__(256) void attn_kernel(
    const unsigned short* __restrict__ Qg, const unsigned short* __restrict__ Kg,
    const unsigned short* __restrict__ Vg, unsigned short* __restrict__ Og) {
  __shared__ __align__(16) unsigned short Klds[2][64 * 72];   // [key][hd]
  __shared__ __align__(16) unsigned short Vlds[2][64 * 72];   // [hd][key] transposed
  __shared__ __align__(16) unsigned short Plds[8 * 16 * 72];  // wave-private P rows
  const int tid = threadIdx.x;
  const int lane = tid & 31;
  const int w = tid >> 5;  // wave 0..7
  const int l15 = lane & 15;
  const int hi = (lane >> 4) & 1;
  const int bh = blockIdx.y;
  const int q0 = blockIdx.x * 128;
  const int b = bh / NH, h = bh % NH;

  // Q fragments (16 rows x 64 hd) loaded once from global
  FragAB qf[2];
  {
    const unsigned short* qp = Qg + ((size_t)bh * SEQ + q0 + w * 16 + l15) * HD;
    int kbase = hi * 8;
#pragma unroll
    for (int ks = 0; ks < 2; ++ks) {
      qf[ks].q[0] = *(const u32x4*)(const void*)&qp[ks * 32 + kbase];
      qf[ks].q[1] = *(const u32x4*)(const void*)&qp[ks * 32 + kbase + 16];
    }
  }

  auto stageKV = [&](int buf, int kb) {
    const unsigned int* kg =
        (const unsigned int*)(Kg + ((size_t)bh * SEQ + kb * 64) * HD);
    unsigned int* Ku = (unsigned int*)Klds[buf];
#pragma unroll
    for (int e = 0; e < 8; ++e) {
      int idx = e * 256 + tid;
      int r = idx >> 5, c = idx & 31;
      Ku[r * 36 + c] = kg[r * 32 + c];
    }
    const unsigned short* vg = Vg + ((size_t)bh * SEQ + kb * 64) * HD;
#pragma unroll
    for (int e = 0; e < 16; ++e) {
      int idx = e * 256 + tid;
      int key = idx >> 6, hd = idx & 63;
      Vlds[buf][hd * 72 + key] = vg[key * 64 + hd];
    }
  };

  f32x8 o[4] = {};
  float mrow[8], lrow[8];
#pragma unroll
  for (int vi = 0; vi < 8; ++vi) { mrow[vi] = -1e30f; lrow[vi] = 0.f; }

  stageKV(0, 0);
  __syncthreads();
  int buf = 0;
  const int rowoff = hi * 8;

  for (int kb = 0; kb < SEQ / 64; ++kb) {
    if (kb + 1 < SEQ / 64) stageKV(buf ^ 1, kb + 1);
    const unsigned short* Kb = Klds[buf];
    const unsigned short* Vb = Vlds[buf];

    // scores S = Q @ K^T  (4 n-tiles x 2 k-steps)
    f32x8 s[4] = {};
#pragma unroll
    for (int ks = 0; ks < 2; ++ks)
#pragma unroll
      for (int ni = 0; ni < 4; ++ni) {
        int off = (ni * 16 + l15) * 72 + ks * 32 + hi * 8;
        FragAB kf;
        kf.q[0] = *(const u32x4*)(const void*)&Kb[off];
        kf.q[1] = *(const u32x4*)(const void*)&Kb[off + 16];
        s[ni] = __builtin_amdgcn_wmma_f32_16x16x32_bf16(
            false, qf[ks].v, false, kf.v, (short)0, s[ni], false, false);
      }

#pragma unroll
    for (int ni = 0; ni < 4; ++ni)
#pragma unroll
      for (int vi = 0; vi < 8; ++vi) s[ni][vi] = s[ni][vi] * 0.125f;

    // online softmax (row = vi + hi*8; stats across 16-lane half-wave)
    float alpha[8];
#pragma unroll
    for (int vi = 0; vi < 8; ++vi) {
      float mx = fmaxf(fmaxf(s[0][vi], s[1][vi]), fmaxf(s[2][vi], s[3][vi]));
      mx = fmaxf(mx, __shfl_xor(mx, 1, 32));
      mx = fmaxf(mx, __shfl_xor(mx, 2, 32));
      mx = fmaxf(mx, __shfl_xor(mx, 4, 32));
      mx = fmaxf(mx, __shfl_xor(mx, 8, 32));
      float mn = fmaxf(mrow[vi], mx);
      alpha[vi] = __expf(mrow[vi] - mn);
      mrow[vi] = mn;
    }
#pragma unroll
    for (int vi = 0; vi < 8; ++vi) {
      float rs = 0.f;
#pragma unroll
      for (int ni = 0; ni < 4; ++ni) {
        float p = __expf(s[ni][vi] - mrow[vi]);
        rs += p;
        Plds[(w * 16 + rowoff + vi) * 72 + ni * 16 + l15] = f2bf(p);
      }
      rs += __shfl_xor(rs, 1, 32);
      rs += __shfl_xor(rs, 2, 32);
      rs += __shfl_xor(rs, 4, 32);
      rs += __shfl_xor(rs, 8, 32);
      lrow[vi] = lrow[vi] * alpha[vi] + rs;
#pragma unroll
      for (int t = 0; t < 4; ++t) o[t][vi] *= alpha[vi];
    }

    // O += P @ V   (P region is wave-private: no block barrier needed)
#pragma unroll
    for (int ks = 0; ks < 2; ++ks) {
      FragAB pf;
      int poff = (w * 16 + l15) * 72 + ks * 32 + hi * 8;
      pf.q[0] = *(const u32x4*)(const void*)&Plds[poff];
      pf.q[1] = *(const u32x4*)(const void*)&Plds[poff + 16];
#pragma unroll
      for (int t = 0; t < 4; ++t) {
        int voff = (t * 16 + l15) * 72 + ks * 32 + hi * 8;
        FragAB vf;
        vf.q[0] = *(const u32x4*)(const void*)&Vb[voff];
        vf.q[1] = *(const u32x4*)(const void*)&Vb[voff + 16];
        o[t] = __builtin_amdgcn_wmma_f32_16x16x32_bf16(
            false, pf.v, false, vf.v, (short)0, o[t], false, false);
      }
    }
    __syncthreads();
    buf ^= 1;
  }

  // finalize: O / l -> bf16 attention output, [B,S,D] layout
#pragma unroll
  for (int vi = 0; vi < 8; ++vi) {
    int qrow = q0 + w * 16 + rowoff + vi;
    float inv = 1.0f / lrow[vi];
#pragma unroll
    for (int t = 0; t < 4; ++t) {
      int col = h * HD + t * 16 + l15;
      Og[((size_t)b * SEQ + qrow) * EMBED + col] = f2bf(o[t][vi] * inv);
    }
  }
}

extern "C" void kernel_launch(void* const* d_in, const int* in_sizes, int n_in,
                              void* d_out, int out_size, void* d_ws, size_t ws_size,
                              hipStream_t stream) {
  const float* x = (const float*)d_in[0];
  const float* Wq = (const float*)d_in[1];
  const float* Wk = (const float*)d_in[2];
  const float* Wv = (const float*)d_in[3];
  const float* Wo = (const float*)d_in[4];
  float* out = (float*)d_out;

  const size_t ACT = (size_t)MTOT * EMBED;   // 3145728 halves
  const size_t WSZ = (size_t)EMBED * EMBED;  // 589824 halves
  unsigned short* xb = (unsigned short*)d_ws;
  unsigned short* qb = xb + ACT;
  unsigned short* kb = qb + ACT;
  unsigned short* vb = kb + ACT;
  unsigned short* ab = vb + ACT;
  unsigned short* wqt = ab + ACT;
  unsigned short* wkt = wqt + WSZ;
  unsigned short* wvt = wkt + WSZ;
  unsigned short* wot = wvt + WSZ;

  int n2 = (int)(ACT / 2);
  cvt_x_kernel<<<(n2 + 255) / 256, 256, 0, stream>>>((const float2*)x,
                                                     (unsigned int*)xb, n2);
  dim3 gt(EMBED / 32, EMBED / 32);
  cvt_wt_kernel<<<gt, 256, 0, stream>>>(Wq, wqt);
  cvt_wt_kernel<<<gt, 256, 0, stream>>>(Wk, wkt);
  cvt_wt_kernel<<<gt, 256, 0, stream>>>(Wv, wvt);
  cvt_wt_kernel<<<gt, 256, 0, stream>>>(Wo, wot);

  dim3 gg(EMBED / 128, MTOT / 128);
  gemm_kernel<0><<<gg, 256, 0, stream>>>(xb, wqt, qb, nullptr);
  gemm_kernel<0><<<gg, 256, 0, stream>>>(xb, wkt, kb, nullptr);
  gemm_kernel<0><<<gg, 256, 0, stream>>>(xb, wvt, vb, nullptr);

  dim3 ga(SEQ / 128, NB * NH);
  attn_kernel<<<ga, 256, 0, stream>>>(qb, kb, vb, ab);

  gemm_kernel<1><<<gg, 256, 0, stream>>>(ab, wot, nullptr, out);
}
